// WMSA_17772574671442
// MI455X (gfx1250) — matrix operand order repaired and checked by reading, not compiled, not run on hardware
//
#include <hip/hip_runtime.h>
#include <hip/hip_bf16.h>

typedef __attribute__((ext_vector_type(16))) _Float16 v16h;
typedef __attribute__((ext_vector_type(8)))  _Float16 v8h;
typedef __attribute__((ext_vector_type(4)))  _Float16 v4h;
typedef __attribute__((ext_vector_type(8)))  float    v8f;
typedef __attribute__((ext_vector_type(4)))  float    v4f;

#define C_    192
#define NQKV  576
#define WA    64
#define HEADS 6
#define HD    32

// LDS strides in halves: multiples of 8 (16B alignment for ds_load_b128) and
// bank-conflict-free for 16-row fragment reads.
#define XS 200   // x-tile / attn-output tile [64][192] (+8 pad)
#define QS 40    // q,k: [head][64 tok][32 dim] (+8 pad)
#define VS 72    // v^T: [head][32 dim][64 tok] (+8 pad)
#define PS 72    // probs: [head][64 q][64 k] (+8 pad)

// ---- WMMA fragment loaders (CDNA5 16-bit layouts, CDNA5 ISA §7.12.2) ----

// A fragment: [M][K] row-major, 16x32 tile starting at p0.
// lane<16: row=lane,    halves0-7 = K0..7,  halves8-15 = K16..23
// lane>=16: row=lane-16, halves0-7 = K8..15, halves8-15 = K24..31
__device__ __forceinline__ v16h frag_a(const _Float16* p0, int strideH, int lane) {
  int r  = lane & 15;
  int kh = (lane & 16) ? 8 : 0;
  const _Float16* p = p0 + r * strideH + kh;
  v8h lo = *(const v8h*)(p);
  v8h hi = *(const v8h*)(p + 16);
  v16h a;
#pragma unroll
  for (int i = 0; i < 8; ++i) { a[i] = lo[i]; a[i + 8] = hi[i]; }
  return a;
}

// B fragment: stored as B^T = [N][K] row-major (K contiguous), 32x16 tile.
// lane<16: col=lane, halves0-7 = K0..7, halves8-15 = K16..23 ; lane>=16: col=lane-16, halves0-7 = K8..15, halves8-15 = K24..31
__device__ __forceinline__ v16h frag_b(const _Float16* p0, int strideH, int lane) {
  int n  = lane & 15;
  int k0 = (lane & 16) ? 8 : 0;
  const _Float16* p = p0 + n * strideH + k0;
  v8h lo = *(const v8h*)(p);
  v8h hi = *(const v8h*)(p + 16);
  v16h b;
#pragma unroll
  for (int i = 0; i < 8; ++i) { b[i] = lo[i]; b[i + 8] = hi[i]; }
  return b;
}

__device__ __forceinline__ v8f wmma_f16(v16h a, v16h b, v8f c) {
  return __builtin_amdgcn_wmma_f32_16x16x32_f16(false, a, false, b, (short)0, c,
                                                false, false);
}

// ---- Prep: f16-transpose weights + precompute rel-pos bias table ----
__global__ void wmsa_prep(const float* __restrict__ wqkv, const float* __restrict__ wout,
                          const float* __restrict__ relp,
                          _Float16* __restrict__ wqkv_t, _Float16* __restrict__ wout_t,
                          float* __restrict__ bias) {
  int idx = blockIdx.x * 256 + threadIdx.x;
  if (idx < NQKV * C_) {                 // wqkv_t[n][k] = wqkv[k][n]
    int n = idx / C_, k = idx % C_;
    wqkv_t[idx] = (_Float16)wqkv[k * NQKV + n];
  }
  int i2 = idx - NQKV * C_;
  if (i2 >= 0 && i2 < C_ * C_) {         // wout_t[n][k] = wout[k][n]
    int n = i2 / C_, k = i2 % C_;
    wout_t[i2] = (_Float16)wout[k * C_ + n];
  }
  int i3 = idx - (NQKV * C_ + C_ * C_);
  if (i3 >= 0 && i3 < HEADS * WA * WA) { // bias[h][q][k]
    int h = i3 >> 12, q = (i3 >> 6) & 63, k = i3 & 63;
    int r0 = (q >> 3) - (k >> 3) + 7;
    int r1 = (q & 7) - (k & 7) + 7;
    bias[i3] = relp[h * 225 + r0 * 15 + r1];
  }
}

// ---- Fused per-window W-MSA: one block = one 8x8 window ----
// LDS = 25600 + 61440 (q/k union probs) + 27648 = 114688 B -> 2 blocks/WGP.
__launch_bounds__(384, 2)
__global__ void wmsa_fused(const float* __restrict__ x,
                           const _Float16* __restrict__ wqkv_t,
                           const float* __restrict__ bqkv,
                           const _Float16* __restrict__ wout_t,
                           const float* __restrict__ bout,
                           const float* __restrict__ bias,
                           float* __restrict__ out) {
  __shared__ alignas(16) _Float16 xo[WA * XS];       // x tile, then attn output O
  __shared__ alignas(16) union QKP {
    struct { _Float16 q[HEADS * WA * QS]; _Float16 k[HEADS * WA * QS]; } qk;
    _Float16 p[HEADS * WA * PS];                     // probs reuse q/k space
  } u;
  __shared__ alignas(16) _Float16 vs[HEADS * HD * VS]; // v transposed [dim][tok]

  const int tid  = threadIdx.x;
  const int lane = tid & 31;
  const int wv   = tid >> 5;                 // 0..11
  const int blk  = blockIdx.x;               // bi*1024 + wi
  const int bi   = blk >> 10;
  const int wi   = blk & 1023;
  const int why  = wi >> 5, wwx = wi & 31;
  const bool masked = (wi == 1023);          // only last window is masked
  const int ln = lane & 15;
  const int m0 = (lane & 16) ? 8 : 0;

  // Phase A: gather rolled x tile (64 tok x 192 ch) f32 -> f16 LDS
  {
    int t0 = tid / 48, c4 = tid % 48;        // exact: 384 = 8*48
#pragma unroll
    for (int j = 0; j < 8; ++j) {
      int t  = t0 + j * 8;
      int gy = (why * 8 + (t >> 3) + 4) & 255; // roll(-4) on read == roll(+4) on write
      int gx = (wwx * 8 + (t & 7) + 4) & 255;
      v4f xv = *(const v4f*)(x + (((size_t)bi * 256 + gy) * 256 + gx) * C_ + c4 * 4);
      v4h hv;
#pragma unroll
      for (int i = 0; i < 4; ++i) hv[i] = (_Float16)xv[i];
      *(v4h*)(&xo[t * XS + c4 * 4]) = hv;
    }
  }
  __syncthreads();

  // Phase B: QKV GEMM (64x192)@(192x576); wave wv owns N-tiles 3wv..3wv+2, all 4 M-tiles
#pragma unroll 1
  for (int ntl = 0; ntl < 3; ++ntl) {
    int nt  = wv * 3 + ntl;
    int n0  = nt * 16;
    int sec = nt / 12;                       // 0=q 1=k 2=v (uniform per wave)
    int hh  = (nt % 12) >> 1;
    int d   = ((nt & 1) << 4) + ln;
    float bv = bqkv[n0 + ln];
#pragma unroll 1
    for (int mt = 0; mt < 4; ++mt) {
      v8f acc;
#pragma unroll
      for (int r = 0; r < 8; ++r) acc[r] = bv;
#pragma unroll
      for (int kk = 0; kk < 6; ++kk) {
        v16h a = frag_a(xo + mt * 16 * XS + kk * 32, XS, lane);
        v16h b = frag_b(wqkv_t + (size_t)n0 * C_ + kk * 32, C_, lane);
        acc = wmma_f16(a, b, acc);
      }
      if (sec == 0) {
#pragma unroll
        for (int r = 0; r < 8; ++r)
          u.qk.q[hh * WA * QS + (mt * 16 + m0 + r) * QS + d] =
              (_Float16)(acc[r] * 0.17677669529663687f);   // HD^-0.5
      } else if (sec == 1) {
#pragma unroll
        for (int r = 0; r < 8; ++r)
          u.qk.k[hh * WA * QS + (mt * 16 + m0 + r) * QS + d] = (_Float16)acc[r];
      } else {
#pragma unroll
        for (int r = 0; r < 8; ++r)
          vs[hh * HD * VS + d * VS + (mt * 16 + m0 + r)] = (_Float16)acc[r];
      }
    }
  }
  __syncthreads();

  // Phase C1: scores + softmax, probs held in registers (2 waves per head,
  // each wave owns 32 query rows = 2 M-tiles).
  const int h    = wv >> 1;
  const int half = wv & 1;
  v8f sp[8];                                  // [mti][nt] normalized probs
  {
    const _Float16* qh  = u.qk.q + h * WA * QS;
    const _Float16* kh_ = u.qk.k + h * WA * QS;
    const float*    bh  = bias + h * WA * WA;
#pragma unroll
    for (int mti = 0; mti < 2; ++mti) {
      int mt = half * 2 + mti;
      v16h aq = frag_a(qh + mt * 16 * QS, QS, lane);     // K=32 == HD, one chunk
      v8f* s = &sp[mti * 4];
#pragma unroll
      for (int nt = 0; nt < 4; ++nt) {
        v8f c0;
#pragma unroll
        for (int r = 0; r < 8; ++r)
          c0[r] = bh[(mt * 16 + m0 + r) * WA + nt * 16 + ln];
        v16h bk = frag_b(kh_ + nt * 16 * QS, QS, lane);
        s[nt] = wmma_f16(aq, bk, c0);
      }
      if (masked) {
#pragma unroll
        for (int nt = 0; nt < 4; ++nt) {
          int n = nt * 16 + ln;
#pragma unroll
          for (int r = 0; r < 8; ++r) {
            int m = mt * 16 + m0 + r;
            if ((m < 4) != (n < 4)) s[nt][r] = -__builtin_inff();
          }
        }
      }
      // row softmax: row m lives in one 16-lane half across the 4 accs
#pragma unroll
      for (int r = 0; r < 8; ++r) {
        float mx = fmaxf(fmaxf(s[0][r], s[1][r]), fmaxf(s[2][r], s[3][r]));
#pragma unroll
        for (int off = 1; off < 16; off <<= 1) mx = fmaxf(mx, __shfl_xor(mx, off, 32));
        float e0 = __expf(s[0][r] - mx), e1 = __expf(s[1][r] - mx);
        float e2 = __expf(s[2][r] - mx), e3 = __expf(s[3][r] - mx);
        float sum = e0 + e1 + e2 + e3;
#pragma unroll
        for (int off = 1; off < 16; off <<= 1) sum += __shfl_xor(sum, off, 32);
        float rinv = 1.0f / sum;
        s[0][r] = e0 * rinv; s[1][r] = e1 * rinv;
        s[2][r] = e2 * rinv; s[3][r] = e3 * rinv;
      }
    }
  }
  __syncthreads();   // q/k fully consumed -> their LDS becomes the probs buffer

  // Phase C2: spill probs (own rows only) to LDS, P(16x64)@V(64x32), O -> xo
  {
    _Float16*       ph = u.p + h * WA * PS;
    const _Float16* vh = vs + h * HD * VS;
#pragma unroll
    for (int mti = 0; mti < 2; ++mti) {
      int mt = half * 2 + mti;
      v8f* s = &sp[mti * 4];
#pragma unroll
      for (int nt = 0; nt < 4; ++nt)
#pragma unroll
        for (int r = 0; r < 8; ++r)
          ph[(mt * 16 + m0 + r) * PS + nt * 16 + ln] = (_Float16)s[nt][r];

      // reads rows this wave just wrote (same-wave LDS ops are in-order)
#pragma unroll
      for (int nt2 = 0; nt2 < 2; ++nt2) {
        v8f o;
#pragma unroll
        for (int r = 0; r < 8; ++r) o[r] = 0.0f;
#pragma unroll
        for (int kc = 0; kc < 2; ++kc) {
          v16h ap = frag_a(ph + mt * 16 * PS + kc * 32, PS, lane);
          v16h bv = frag_b(vh + nt2 * 16 * VS + kc * 32, VS, lane);
          o = wmma_f16(ap, bv, o);
        }
#pragma unroll
        for (int r = 0; r < 8; ++r)
          xo[(mt * 16 + m0 + r) * XS + h * 32 + nt2 * 16 + ln] = (_Float16)o[r];
      }
    }
  }
  __syncthreads();

  // Phase D: output projection (64x192)@(192x192) + bias, rolled store
  {
    int n0 = wv * 16;                        // 12 waves x 16 = 192 cols
    float bv = bout[n0 + ln];
#pragma unroll 1
    for (int mt = 0; mt < 4; ++mt) {
      v8f acc;
#pragma unroll
      for (int r = 0; r < 8; ++r) acc[r] = bv;
#pragma unroll
      for (int kk = 0; kk < 6; ++kk) {
        v16h a = frag_a(xo + mt * 16 * XS + kk * 32, XS, lane);
        v16h b = frag_b(wout_t + (size_t)n0 * C_ + kk * 32, C_, lane);
        acc = wmma_f16(a, b, acc);
      }
#pragma unroll
      for (int r = 0; r < 8; ++r) {
        int m  = mt * 16 + m0 + r;
        int gy = (why * 8 + (m >> 3) + 4) & 255;
        int gx = (wwx * 8 + (m & 7) + 4) & 255;
        out[(((size_t)bi * 256 + gy) * 256 + gx) * C_ + n0 + ln] = acc[r];
      }
    }
  }
}

extern "C" void kernel_launch(void* const* d_in, const int* in_sizes, int n_in,
                              void* d_out, int out_size, void* d_ws, size_t ws_size,
                              hipStream_t stream) {
  const float* x    = (const float*)d_in[0];
  const float* wqkv = (const float*)d_in[1];
  const float* bqkv = (const float*)d_in[2];
  const float* relp = (const float*)d_in[3];
  const float* wout = (const float*)d_in[4];
  const float* bout = (const float*)d_in[5];
  float* out = (float*)d_out;

  char* ws = (char*)d_ws;
  _Float16* wqkv_t = (_Float16*)ws;                       // 576*192*2  = 221184 B
  _Float16* wout_t = (_Float16*)(ws + 221184);            // 192*192*2  =  73728 B
  float*    bias   = (float*)(ws + 221184 + 73728);       // 6*64*64*4  =  98304 B

  // 110592 + 36864 + 24576 = 172032 work items
  wmsa_prep<<<672, 256, 0, stream>>>(wqkv, wout, relp, wqkv_t, wout_t, bias);
  // 4 batches * 1024 windows; 12 waves (384 threads) per window
  wmsa_fused<<<4096, 384, 0, stream>>>(x, wqkv_t, bqkv, wout_t, bout, bias, out);
}
